// BiLSTMCRF_86715389706663
// MI455X (gfx1250) — compile-verified, matrix operand order checked by reading, not run on hardware
//
#include <hip/hip_runtime.h>

// ---------------------------------------------------------------------------
// BiLSTM-CRF for MI455X (gfx1250), wave32 + WMMA bf16.
// B=64, T=256, E=256, HD=256 (4HD=1024 gates), H=512, TAGS=20, M=T*B=16384.
// ---------------------------------------------------------------------------

typedef __bf16 bf16_t;
typedef __attribute__((ext_vector_type(16))) __bf16 v16bf;
typedef __attribute__((ext_vector_type(8)))  __bf16 v8bf;
typedef __attribute__((ext_vector_type(8)))  float   v8f;

#define NB   64
#define NT   256
#define NE   256
#define NHD  256
#define NG   1024          // 4*HD
#define NM   (NT * NB)     // 16384 rows
#define NTAGS 20

// f32 -> bf16 round-to-nearest-even via bit ops (no reliance on native cvt)
__device__ inline bf16_t f2bf(float f) {
  union { float f; unsigned u; } a; a.f = f;
  unsigned r = a.u + 0x7FFFu + ((a.u >> 16) & 1u);
  union { unsigned short s; bf16_t b; } o;
  o.s = (unsigned short)(r >> 16);
  return o.b;
}

__device__ inline float sigf(float x) { return 1.0f / (1.0f + expf(-x)); }

// Load 16 bf16 K-values for one lane of a WMMA A/B operand.
// p points at (row_base + k_chunk + half*8); elements [0..7] = K offsets 0..7,
// elements [8..15] = K offsets 16..23 relative to p-16*half... i.e. p and p+16.
__device__ inline v16bf ld16(const bf16_t* p) {
  union { v16bf v; v8bf h[2]; } u;
  u.h[0] = *(const v8bf*)(p);
  u.h[1] = *(const v8bf*)(p + 16);
  return u.v;
}

// ---------------------------------------------------------------------------
// Kernel 1: embedding gather -> x bf16 [M=16384][256], row = t*B + b
// ---------------------------------------------------------------------------
__global__ void k_embed(const int* __restrict__ sent, const float* __restrict__ emb,
                        bf16_t* __restrict__ x) {
  const int row = blockIdx.x;            // t*64 + b
  const int t = row >> 6, b = row & 63;
  const int tok = sent[b * NT + t];      // sentences is [B,T]
  const int e = threadIdx.x;             // 256 threads
  x[(size_t)row * NE + e] = f2bf(emb[(size_t)tok * NE + e]);
}

// ---------------------------------------------------------------------------
// Kernel 2: f32 -> bf16 matrix convert (weights)
// ---------------------------------------------------------------------------
__global__ void k_cvt(const float* __restrict__ src, bf16_t* __restrict__ dst, int n) {
  int i = blockIdx.x * blockDim.x + threadIdx.x;
  if (i < n) dst[i] = f2bf(src[i]);
}

// ---------------------------------------------------------------------------
// Kernel 3: bias_sum[dir][g] = bih[g] + bhh[g]
// ---------------------------------------------------------------------------
__global__ void k_bias(const float* __restrict__ bihf, const float* __restrict__ bhhf,
                       const float* __restrict__ bihb, const float* __restrict__ bhhb,
                       float* __restrict__ bias) {
  int i = blockIdx.x * blockDim.x + threadIdx.x;   // 0..2047
  if (i < NG)            bias[i]      = bihf[i]        + bhhf[i];
  else if (i < 2 * NG)   bias[i]      = bihb[i - NG]   + bhhb[i - NG];
}

// ---------------------------------------------------------------------------
// Kernel 4: input projection GEMM  gx[dir] = x @ Wih[dir]^T   (no bias)
//   M=16384, K=256, N=1024.  One wave = one 16x16 output tile, 8 WMMA along K.
//   grid = (1024 Mtiles, 8 Ntile-groups, 2 dirs), block = 256 (8 waves).
// ---------------------------------------------------------------------------
__global__ void k_gemm_in(const bf16_t* __restrict__ x,    // [16384][256]
                          const bf16_t* __restrict__ wih,  // [2][1024][256]
                          float* __restrict__ gx) {        // [2][16384][1024]
  const int tid  = threadIdx.x;
  const int lane = tid & 31, wave = tid >> 5;
  const int mtile = blockIdx.x;
  const int ntile = blockIdx.y * 8 + wave;
  const int dir   = blockIdx.z;
  const int n16  = lane & 15;          // A: M row within tile; B: N col within tile
  const int half = lane >> 4;

  const bf16_t* xr = x   + (size_t)(mtile * 16 + n16) * NE + half * 8;
  const bf16_t* wr = wih + (size_t)dir * NG * NE
                         + (size_t)(ntile * 16 + n16) * NE + half * 8;
  v8f acc = {};
#pragma unroll
  for (int k = 0; k < 8; ++k) {
    v16bf a = ld16(xr + k * 32);
    v16bf b = ld16(wr + k * 32);
    acc = __builtin_amdgcn_wmma_f32_16x16x32_bf16(false, a, false, b,
                                                  (short)0, acc, false, false);
  }
  float* out = gx + (size_t)dir * NM * NG;
#pragma unroll
  for (int r = 0; r < 8; ++r)
    out[(size_t)(mtile * 16 + r + 8 * half) * NG + ntile * 16 + n16] = acc[r];
}

// ---------------------------------------------------------------------------
// Kernel 5: recurrent LSTM (both directions).
//   grid = 8 persistent WGs: blockIdx = dir*4 + batch_group(16 rows).
//   block = 512 (16 waves). Each step: g = h @ Whh^T via WMMA (64 Ntiles,
//   4 per wave), stage in LDS gbuf, then elementwise gates update h,c.
//   Dynamic LDS: gbuf f32 16x1024 (64KB) + cbuf f32 16x256 (16KB)
//                + hbuf bf16 16x256 (8KB) = 90112 B (legal: 320KB/WGP).
// ---------------------------------------------------------------------------
__global__ void k_lstm(const float* __restrict__ gx,    // [2][M][1024]
                       const bf16_t* __restrict__ whh,  // [2][1024][256] bf16
                       const float* __restrict__ bias,  // [2][1024]
                       const float* __restrict__ h0,    // [2][64][256]
                       const float* __restrict__ c0,    // [2][64][256]
                       float* __restrict__ hcat) {      // [M][512] (fused concat)
  extern __shared__ char smem[];
  float*  gbuf = (float*)smem;                               // [16][1024]
  float*  cbuf = (float*)(smem + 16 * NG * 4);               // [16][256]
  bf16_t* hbuf = (bf16_t*)(smem + 16 * NG * 4 + 16 * NHD * 4); // [16][256]

  const int dir = blockIdx.x >> 2;
  const int bg  = blockIdx.x & 3;
  const int tid = threadIdx.x;
  const int lane = tid & 31, wave = tid >> 5;
  const int bbase = bg * 16;

  const float*  gxd   = gx   + (size_t)dir * NM * NG;
  const bf16_t* whhd  = whh  + (size_t)dir * NG * NHD;
  const float*  biasd = bias + dir * NG;

  // init h (bf16) and c (f32) state
  for (int u = tid; u < 16 * NHD; u += blockDim.x) {
    int b = u >> 8, j = u & 255;
    int gi = dir * (NB * NHD) + (bbase + b) * NHD + j;
    hbuf[u] = f2bf(h0[gi]);
    cbuf[u] = c0[gi];
  }
  __syncthreads();

  const int n16  = lane & 15;
  const int half = lane >> 4;

  for (int step = 0; step < NT; ++step) {
    const int t = dir ? (NT - 1 - step) : step;

    // prefetch next step's precomputed input-projection rows into caches
    if (step + 1 < NT) {
      const int tn = dir ? (NT - 2 - step) : (step + 1);
      const float* pf = gxd + (size_t)(tn * NB + bbase + (tid >> 5)) * NG
                            + (size_t)(tid & 31) * 32;
      __builtin_prefetch(pf, 0, 1);
    }

    // ---- Phase A: gbuf[b][g] = sum_k hbuf[b][k] * Whh[g][k] (WMMA) ----
    {
      v16bf a[8];
#pragma unroll
      for (int k = 0; k < 8; ++k)
        a[k] = ld16(hbuf + n16 * NHD + k * 32 + half * 8);   // LDS ds_load_b128

#pragma unroll
      for (int nt = 0; nt < 4; ++nt) {
        const int ntile = wave * 4 + nt;
        const bf16_t* wr = whhd + (size_t)(ntile * 16 + n16) * NHD + half * 8;
        v8f acc = {};
#pragma unroll
        for (int k = 0; k < 8; ++k) {
          v16bf b = ld16(wr + k * 32);
          acc = __builtin_amdgcn_wmma_f32_16x16x32_bf16(false, a[k], false, b,
                                                        (short)0, acc, false, false);
        }
#pragma unroll
        for (int r = 0; r < 8; ++r)
          gbuf[(r + 8 * half) * NG + ntile * 16 + n16] = acc[r];
      }
    }
    __syncthreads();

    // ---- Phase B: elementwise gate math (PyTorch order i,f,g,o) ----
    for (int u = tid; u < 16 * NHD; u += blockDim.x) {
      const int b = u >> 8, j = u & 255;
      const size_t row = (size_t)(t * NB + bbase + b);
      const float* gxr = gxd + row * NG;
      const float gi = gbuf[b * NG + j]            + gxr[j]            + biasd[j];
      const float gf = gbuf[b * NG + NHD + j]      + gxr[NHD + j]      + biasd[NHD + j];
      const float gg = gbuf[b * NG + 2 * NHD + j]  + gxr[2 * NHD + j]  + biasd[2 * NHD + j];
      const float go = gbuf[b * NG + 3 * NHD + j]  + gxr[3 * NHD + j]  + biasd[3 * NHD + j];
      const float c  = sigf(gf) * cbuf[u] + sigf(gi) * tanhf(gg);
      const float h  = sigf(go) * tanhf(c);
      cbuf[u] = c;
      hbuf[u] = f2bf(h);
      hcat[row * 512 + dir * NHD + j] = h;   // fused [hf|hb] concat
    }
    __syncthreads();
  }
}

// ---------------------------------------------------------------------------
// Kernel 6: logits[r][tag] = hcat[r] . Wl[tag] + bl[tag]   (168 MFLOP, VALU)
// ---------------------------------------------------------------------------
__global__ void k_logits(const float* __restrict__ hcat, const float* __restrict__ Wl,
                         const float* __restrict__ bl, float* __restrict__ logits) {
  __shared__ float row[512];
  const int r = blockIdx.x;
  for (int i = threadIdx.x; i < 512; i += blockDim.x)
    row[i] = hcat[(size_t)r * 512 + i];
  __syncthreads();
  const int tg = threadIdx.x;
  if (tg < NTAGS) {
    const float* w = Wl + tg * 512;
    float s = bl[tg];
#pragma unroll 8
    for (int i = 0; i < 512; ++i) s += row[i] * w[i];
    logits[(size_t)r * NTAGS + tg] = s;
  }
}

// ---------------------------------------------------------------------------
// Kernel 7: Viterbi DP + backtrack + softmax probs. 1 block, 64 threads,
// one batch element per thread (first-max tie-break like jnp.argmax).
// out = [scores(64) | path(64*256) | probs(64*256)] all f32.
// ---------------------------------------------------------------------------
__global__ void k_viterbi(const float* __restrict__ logits,  // [T][B][20]
                          const float* __restrict__ trans,   // [20][20]
                          float* __restrict__ trell,         // [T][B][20] ws
                          int* __restrict__ bps,             // [T][B][20] ws
                          float* __restrict__ out) {
  __shared__ float tr_s[NTAGS * NTAGS];
  const int tid = threadIdx.x;
  for (int i = tid; i < NTAGS * NTAGS; i += blockDim.x) tr_s[i] = trans[i];
  __syncthreads();

  const int b = tid;
  float prev[NTAGS];
#pragma unroll
  for (int k = 0; k < NTAGS; ++k) {
    prev[k] = logits[b * NTAGS + k];
    trell[b * NTAGS + k] = prev[k];
  }
  for (int t = 1; t < NT; ++t) {
    float cur[NTAGS];
#pragma unroll
    for (int c = 0; c < NTAGS; ++c) {
      float best = prev[0] + tr_s[0 * NTAGS + c];
      int bi = 0;
#pragma unroll
      for (int p = 1; p < NTAGS; ++p) {
        float v = prev[p] + tr_s[p * NTAGS + c];
        if (v > best) { best = v; bi = p; }
      }
      const float nv = logits[(size_t)(t * NB + b) * NTAGS + c] + best;
      cur[c] = nv;
      trell[(size_t)(t * NB + b) * NTAGS + c] = nv;
      bps[(size_t)(t * NB + b) * NTAGS + c] = bi;
    }
#pragma unroll
    for (int k = 0; k < NTAGS; ++k) prev[k] = cur[k];
  }

  // scores + last tag (first-max)
  float best = prev[0]; int tag = 0;
#pragma unroll
  for (int k = 1; k < NTAGS; ++k) if (prev[k] > best) { best = prev[k]; tag = k; }
  out[b] = best;

  float* path_o  = out + NB;
  float* probs_o = out + NB + NB * NT;
  int cur_tag = tag;
  for (int t = NT - 1; t >= 0; --t) {
    path_o[(size_t)b * NT + t] = (float)cur_tag;
    const float* tr = trell + (size_t)(t * NB + b) * NTAGS;
    float mx = tr[0];
#pragma unroll
    for (int k = 1; k < NTAGS; ++k) mx = fmaxf(mx, tr[k]);
    float se = 0.f;
#pragma unroll
    for (int k = 0; k < NTAGS; ++k) se += expf(tr[k] - mx);
    probs_o[(size_t)b * NT + t] = expf(tr[cur_tag] - mx) / se;
    if (t > 0) cur_tag = bps[(size_t)(t * NB + b) * NTAGS + cur_tag];
  }
}

// ---------------------------------------------------------------------------
// Host launch
// ---------------------------------------------------------------------------
extern "C" void kernel_launch(void* const* d_in, const int* in_sizes, int n_in,
                              void* d_out, int out_size, void* d_ws, size_t ws_size,
                              hipStream_t stream) {
  const int*   sent  = (const int*)  d_in[0];
  const float* emb   = (const float*)d_in[1];
  const float* Wih_f = (const float*)d_in[2];
  const float* Whh_f = (const float*)d_in[3];
  const float* bih_f = (const float*)d_in[4];
  const float* bhh_f = (const float*)d_in[5];
  const float* Wih_b = (const float*)d_in[6];
  const float* Whh_b = (const float*)d_in[7];
  const float* bih_b = (const float*)d_in[8];
  const float* bhh_b = (const float*)d_in[9];
  const float* Wl    = (const float*)d_in[10];
  const float* bl    = (const float*)d_in[11];
  const float* trans = (const float*)d_in[12];
  const float* h0    = (const float*)d_in[13];
  const float* c0    = (const float*)d_in[14];

  char* ws = (char*)d_ws;
  const size_t SZ_X    = (size_t)NM * NE * 2;       // 8 MB   bf16 x
  const size_t SZ_W    = (size_t)NG * NE * 2;       // 512 KB per weight matrix
  const size_t SZ_BIAS = (size_t)NG * 4;            // 4 KB per dir
  const size_t SZ_GX   = (size_t)NM * NG * 4;       // 64 MB per dir
  const size_t SZ_HCAT = (size_t)NM * 512 * 4;      // 32 MB
  const size_t SZ_LOG  = (size_t)NM * NTAGS * 4;    // 1.25 MB

  bf16_t* x_bf   = (bf16_t*)(ws);
  bf16_t* wih_bf = (bf16_t*)(ws + SZ_X);                       // [2][1024][256]
  bf16_t* whh_bf = (bf16_t*)(ws + SZ_X + 2 * SZ_W);            // [2][1024][256]
  float*  bias   = (float*) (ws + SZ_X + 4 * SZ_W);            // [2][1024]
  float*  gx     = (float*) (ws + SZ_X + 4 * SZ_W + 2 * SZ_BIAS);
  float*  hcat   = (float*) ((char*)gx + 2 * SZ_GX);
  float*  logits = (float*) ((char*)hcat + SZ_HCAT);
  float*  trell  = (float*) ((char*)logits + SZ_LOG);
  int*    bps    = (int*)   ((char*)trell + SZ_LOG);

  // 1. embedding gather -> bf16
  k_embed<<<NM, NE, 0, stream>>>(sent, emb, x_bf);

  // 2. weight conversion to bf16 (dir-major layout)
  const int NW = NG * NE;  // 262144
  k_cvt<<<(NW + 255) / 256, 256, 0, stream>>>(Wih_f, wih_bf,            NW);
  k_cvt<<<(NW + 255) / 256, 256, 0, stream>>>(Wih_b, wih_bf + (size_t)NW, NW);
  k_cvt<<<(NW + 255) / 256, 256, 0, stream>>>(Whh_f, whh_bf,            NW);
  k_cvt<<<(NW + 255) / 256, 256, 0, stream>>>(Whh_b, whh_bf + (size_t)NW, NW);

  // 3. bias sums
  k_bias<<<8, 256, 0, stream>>>(bih_f, bhh_f, bih_b, bhh_b, bias);

  // 4. input projections, both directions (WMMA)
  k_gemm_in<<<dim3(NM / 16, 8, 2), 256, 0, stream>>>(x_bf, wih_bf, gx);

  // 5. recurrent LSTM, both directions (WMMA, persistent WGs, 88KB LDS)
  const size_t lstm_lds = 16 * NG * 4 + 16 * NHD * 4 + 16 * NHD * 2; // 90112
  k_lstm<<<8, 512, lstm_lds, stream>>>(gx, whh_bf, bias, h0, c0, hcat);

  // 6. tag logits
  k_logits<<<NM, 128, 0, stream>>>(hcat, Wl, bl, logits);

  // 7. Viterbi + backtrack + probs -> d_out
  k_viterbi<<<1, 64, 0, stream>>>(logits, trans, trell, bps, (float*)d_out);
}
